// MiniSelfAttention_87256555585606
// MI455X (gfx1250) — matrix operand order repaired and checked by reading, not compiled, not run on hardware
//
#include <hip/hip_runtime.h>
#include <math.h>

// ---------------------------------------------------------------------------
// MiniSelfAttention forward for gfx1250 (MI455X).
//   B=4, T=2048, V=1024, H=16, D=64
//   Stage 0: x, Wq, Wk, Wv, Wo converted to bf16 once (elementwise kernel)
//   Stage 1: q/k/v = x @ W^T   (pure bf16 WMMA GEMM, f32 accum)
//            V stored head-transposed: VT[(b,h,d), t] for fast P@V loads
//   Stage 2: flash attention per (b,h), S streamed in chunks of 32
//            row-sums via WMMA-with-ones, wave-local s_wait_dscnt sync
//   Stage 3: out = attn @ Wo^T + bo  (f32 output)
// ---------------------------------------------------------------------------

typedef __attribute__((ext_vector_type(16))) __bf16 v16bf;
typedef __attribute__((ext_vector_type(8)))  float  v8f;

union BF16x16 {
    v16bf v;
    unsigned u[8];
    unsigned short s[16];
};

// float -> bf16 bits, round-to-nearest-even
__device__ __forceinline__ unsigned short f2bf(float f) {
    unsigned u = __builtin_bit_cast(unsigned, f);
    unsigned r = u + 0x7FFFu + ((u >> 16) & 1u);
    return (unsigned short)(r >> 16);
}
__device__ __forceinline__ unsigned pack2bf(float lo, float hi) {
    return (unsigned)f2bf(lo) | ((unsigned)f2bf(hi) << 16);
}

// A-operand K offset for 16-bit 16x32 A tile (ISA 7.12.2):
//   VGPR j, lane-half hl -> K = 16*(j>>2) + 8*hl + 2*(j&3)  (pair K, K+1)
__device__ __forceinline__ int ka_off(int j, int hl) {
    return 16 * (j >> 2) + 8 * hl + 2 * (j & 3);
}
// B-operand K offset for 16-bit 32x16 B tile (same K order as A):
//   VGPR j, lane-half hl -> K = 16*(j>>2) + 8*hl + 2*(j&3)  (pair K, K+1)
__device__ __forceinline__ int kb_off(int j, int hl) {
    return 16 * (j >> 2) + 8 * hl + 2 * (j & 3);
}

__device__ __forceinline__ v8f wmma_bf16(const BF16x16& a, const BF16x16& b, v8f c) {
    return __builtin_amdgcn_wmma_f32_16x16x32_bf16(
        /*neg_a=*/false, a.v, /*neg_b=*/false, b.v,
        /*c_mod=*/(short)0, c, /*reuse_a=*/false, /*reuse_b=*/false);
}

__device__ __forceinline__ void store_out(float* p, float v) { *p = v; }
__device__ __forceinline__ void store_out(unsigned short* p, float v) { *p = f2bf(v); }

// ---------------------------------------------------------------------------
// Elementwise f32 -> bf16 (8 elements / thread). n must be a multiple of 8.
// ---------------------------------------------------------------------------
__global__ void f32_to_bf16_kernel(const float* __restrict__ src,
                                   unsigned short* __restrict__ dst, long n)
{
    long i = ((long)blockIdx.x * blockDim.x + threadIdx.x) * 8;
    if (i + 8 <= n) {
        float4 a = *(const float4*)(src + i);
        float4 b = *(const float4*)(src + i + 4);
        uint4 o = make_uint4(pack2bf(a.x, a.y), pack2bf(a.z, a.w),
                             pack2bf(b.x, b.y), pack2bf(b.z, b.w));
        *(uint4*)(dst + i) = o;
    }
}

// ---------------------------------------------------------------------------
// Y[M,N] = X[M,K] @ W[N,K]^T (+ bias), all-bf16 operands, f32 accum.
// Block = 256 thr = 8 waves (2x4 wave grid). Block tile 64(M) x 256(N).
// Each wave: 32(M) x 64(N) -> 8 wmma per K=32 step, 12 b128 loads.
// TRANS_V: store head-transposed bf16 into VT[((b*16+h)*64+d)*2048 + t].
// ---------------------------------------------------------------------------
template <typename OT, bool HAS_BIAS, bool TRANS_V>
__global__ void gemm_bf16_kernel(const unsigned short* __restrict__ X,
                                 const unsigned short* __restrict__ W,
                                 const float* __restrict__ bias,
                                 OT* __restrict__ Y, int M, int N, int K)
{
    const int tid  = threadIdx.x;
    const int wid  = tid >> 5;
    const int lane = tid & 31;
    const int ln   = lane & 15;
    const int hl   = lane >> 4;
    const int wm   = wid >> 2;      // 0..1
    const int wn   = wid & 3;       // 0..3
    const int m0   = blockIdx.y * 64 + wm * 32;
    const int n0   = blockIdx.x * 256 + wn * 64;

    v8f zf = {0.f, 0.f, 0.f, 0.f, 0.f, 0.f, 0.f, 0.f};
    v8f acc[2][4] = {{zf, zf, zf, zf}, {zf, zf, zf, zf}};

    const unsigned short* xr0 = X + (long)(m0 + ln) * K;
    const unsigned short* xr1 = X + (long)(m0 + 16 + ln) * K;

    for (int k0 = 0; k0 < K; k0 += 32) {
        __builtin_prefetch(xr0 + k0 + 256, 0, 1);   // global_prefetch_b8
        BF16x16 a0, a1;
#pragma unroll
        for (int j = 0; j < 8; ++j) {
            const int kk = k0 + ka_off(j, hl);
            a0.u[j] = *(const unsigned*)(xr0 + kk);
            a1.u[j] = *(const unsigned*)(xr1 + kk);
        }
#pragma unroll
        for (int t = 0; t < 4; ++t) {
            const unsigned short* brow = W + (long)(n0 + t * 16 + ln) * K;
            BF16x16 bm;
#pragma unroll
            for (int j = 0; j < 8; ++j)
                bm.u[j] = *(const unsigned*)(brow + k0 + kb_off(j, hl));
            acc[0][t] = wmma_bf16(a0, bm, acc[0][t]);
            acc[1][t] = wmma_bf16(a1, bm, acc[1][t]);
        }
    }

    // C/D layout: vgpr r, lane -> (m = r + 8*hl, n = ln)
#pragma unroll
    for (int im = 0; im < 2; ++im) {
#pragma unroll
        for (int t = 0; t < 4; ++t) {
            const int col = n0 + t * 16 + ln;
            const float bv = HAS_BIAS ? bias[col] : 0.0f;
#pragma unroll
            for (int r = 0; r < 8; ++r) {
                const int row = m0 + im * 16 + 8 * hl + r;
                const float val = acc[im][t][r] + bv;
                if (TRANS_V) {
                    // row = b*2048 + t_pos ; col = h*64 + d
                    const int bb = row >> 11, tt = row & 2047;
                    const int head = col >> 6, d = col & 63;
                    Y[(((long)bb * 16 + head) * 64 + d) * 2048 + tt] = (OT)f2bf(val);
                } else {
                    store_out(Y + (long)row * N + col, val);
                }
            }
        }
    }
}

// ---------------------------------------------------------------------------
// Flash attention: bf16 Q/K (row layout), bf16 VT (head-transposed), bf16 out.
// Grid: (T/128, H, B), 256 thr. Each wave: 16 query rows x D=64, S chunks of 32.
// Softmax: per-8-row-group running max (valid: any m >= rowmax works),
// per-row normalizer ls via WMMA row-sum against an all-ones B operand.
// LDS exchange is wave-private -> s_wait_dscnt instead of block barriers.
// ---------------------------------------------------------------------------
__global__ void attn_fwd_kernel(const unsigned short* __restrict__ Q,
                                const unsigned short* __restrict__ Kb,
                                const unsigned short* __restrict__ VT,
                                unsigned short* __restrict__ O)
{
    constexpr int T = 2048, V = 1024;
    __shared__ unsigned short pbuf[8][16][32];   // per-wave P tile (bf16)

    const int tid  = threadIdx.x;
    const int wid  = tid >> 5;
    const int lane = tid & 31;
    const int ln   = lane & 15;
    const int hl   = lane >> 4;
    const int b    = blockIdx.z;
    const int h    = blockIdx.y;
    const int t0   = blockIdx.x * 128 + wid * 16;
    const long base = (long)b * T * V + h * 64;

    // Preload Q rows (A operand), split D=64 into two K=32 chunks
    BF16x16 qa[2];
    {
        const unsigned short* qrow = Q + base + (long)(t0 + ln) * V;
#pragma unroll
        for (int c = 0; c < 2; ++c)
#pragma unroll
            for (int j = 0; j < 8; ++j)
                qa[c].u[j] = *(const unsigned*)(qrow + 32 * c + ka_off(j, hl));
    }

    // Constant all-ones B operand (bf16 1.0 = 0x3F80) for row sums
    BF16x16 onesB;
#pragma unroll
    for (int j = 0; j < 8; ++j) onesB.u[j] = 0x3F803F80u;

    v8f zf = {0.f, 0.f, 0.f, 0.f, 0.f, 0.f, 0.f, 0.f};
    v8f oacc[4] = {zf, zf, zf, zf};
    float ls[8];
#pragma unroll
    for (int r = 0; r < 8; ++r) ls[r] = 0.0f;
    float mx = -1.0e30f;   // running group max for this half's 8 rows

    const float scale = 0.125f;   // 1/sqrt(64)

    for (int s0 = 0; s0 < T; s0 += 32) {
        // ---- scores: 16 x 32, two 16-col tiles, K=64 (2 wmma each) ----
        v8f sc[2];
#pragma unroll
        for (int tn = 0; tn < 2; ++tn) {
            const unsigned short* krow = Kb + base + (long)(s0 + tn * 16 + ln) * V;
            BF16x16 kb0, kb1;
#pragma unroll
            for (int j = 0; j < 8; ++j) {
                const int kk = kb_off(j, hl);
                kb0.u[j] = *(const unsigned*)(krow + kk);
                kb1.u[j] = *(const unsigned*)(krow + 32 + kk);
            }
            v8f s = zf;
            s = wmma_bf16(qa[0], kb0, s);
            s = wmma_bf16(qa[1], kb1, s);
            sc[tn] = s;
        }

        // ---- scale + group max over this half's 8 rows x 32 cols ----
        float lm = -1.0e30f;
#pragma unroll
        for (int r = 0; r < 8; ++r) {
            sc[0][r] *= scale;
            sc[1][r] *= scale;
            lm = fmaxf(lm, fmaxf(sc[0][r], sc[1][r]));
        }
        lm = fmaxf(lm, __shfl_xor(lm, 1, 32));
        lm = fmaxf(lm, __shfl_xor(lm, 2, 32));
        lm = fmaxf(lm, __shfl_xor(lm, 4, 32));
        lm = fmaxf(lm, __shfl_xor(lm, 8, 32));
        const float nm = fmaxf(mx, lm);
        const float al = __expf(mx - nm);
        mx = nm;

        // ---- exponentials + rescale accumulators ----
#pragma unroll
        for (int r = 0; r < 8; ++r) {
            sc[0][r] = __expf(sc[0][r] - nm);
            sc[1][r] = __expf(sc[1][r] - nm);
            ls[r] *= al;
#pragma unroll
            for (int t = 0; t < 4; ++t) oacc[t][r] *= al;
        }

        // ---- C-layout -> A-layout transpose of P through LDS ----
        // Wave-private region: DS ops of one wave complete in order, so a
        // dscnt drain makes all lanes' stores visible to all lanes' loads.
#pragma unroll
        for (int r = 0; r < 8; ++r) {
            const int row = 8 * hl + r;
            pbuf[wid][row][ln]      = f2bf(sc[0][r]);
            pbuf[wid][row][16 + ln] = f2bf(sc[1][r]);
        }
        asm volatile("s_wait_dscnt 0" ::: "memory");
        BF16x16 pa;
#pragma unroll
        for (int j = 0; j < 8; ++j)
            pa.u[j] = *(const unsigned*)&pbuf[wid][ln][ka_off(j, hl)];
        asm volatile("" ::: "memory");   // keep next-iter stores after these loads

        // ---- per-row sums via WMMA against all-ones B ----
        v8f sums = wmma_bf16(pa, onesB, zf);
#pragma unroll
        for (int r = 0; r < 8; ++r) ls[r] += sums[r];

        // ---- oacc += P(16x32) @ V(32x64) using head-transposed V ----
        // VT row for d = t*16+ln holds V[:, h*64+d] contiguously over s.
#pragma unroll
        for (int t = 0; t < 4; ++t) {
            const unsigned short* vtr =
                VT + ((long)(b * 16 + h) * 64 + t * 16 + ln) * T + s0;
            BF16x16 vm;
#pragma unroll
            for (int j = 0; j < 8; ++j)
                vm.u[j] = *(const unsigned*)(vtr + kb_off(j, hl));
            oacc[t] = wmma_bf16(pa, vm, oacc[t]);
        }
    }

    // ---- normalize and store bf16 ----
#pragma unroll
    for (int r = 0; r < 8; ++r) {
        const float inv = 1.0f / ls[r];
        const int row = t0 + 8 * hl + r;
#pragma unroll
        for (int t = 0; t < 4; ++t) {
            const int col = t * 16 + ln;
            O[base + (long)row * V + col] = f2bf(oacc[t][r] * inv);
        }
    }
}

// ---------------------------------------------------------------------------
extern "C" void kernel_launch(void* const* d_in, const int* in_sizes, int n_in,
                              void* d_out, int out_size, void* d_ws, size_t ws_size,
                              hipStream_t stream)
{
    const float* x  = (const float*)d_in[0];
    const float* Wq = (const float*)d_in[1];
    const float* Wk = (const float*)d_in[2];
    const float* Wv = (const float*)d_in[3];
    const float* Wo = (const float*)d_in[4];
    const float* bo = (const float*)d_in[5];
    float* out = (float*)d_out;

    const int Bb = 4, T = 2048, V = 1024;
    const int M = Bb * T, N = V, K = V;
    const long NE = (long)M * V;        // 8388608
    const long WN = (long)V * V;        // 1048576

    unsigned short* p   = (unsigned short*)d_ws;
    unsigned short* xb  = p;  p += NE;
    unsigned short* wqb = p;  p += WN;
    unsigned short* wkb = p;  p += WN;
    unsigned short* wvb = p;  p += WN;
    unsigned short* wob = p;  p += WN;
    unsigned short* qb  = p;  p += NE;
    unsigned short* kb  = p;  p += NE;
    unsigned short* vtb = p;  p += NE;  // head-transposed V
    unsigned short* ab  = p;  p += NE;

    dim3 blk(256);

    // Stage 0: one-time bf16 conversion
    f32_to_bf16_kernel<<<dim3((unsigned)(NE / 2048)), blk, 0, stream>>>(x,  xb,  NE);
    f32_to_bf16_kernel<<<dim3((unsigned)(WN / 2048)), blk, 0, stream>>>(Wq, wqb, WN);
    f32_to_bf16_kernel<<<dim3((unsigned)(WN / 2048)), blk, 0, stream>>>(Wk, wkb, WN);
    f32_to_bf16_kernel<<<dim3((unsigned)(WN / 2048)), blk, 0, stream>>>(Wv, wvb, WN);
    f32_to_bf16_kernel<<<dim3((unsigned)(WN / 2048)), blk, 0, stream>>>(Wo, wob, WN);

    // Stage 1: projections
    dim3 ggrid(N / 256, M / 64);
    gemm_bf16_kernel<unsigned short, false, false><<<ggrid, blk, 0, stream>>>(xb, wqb, nullptr, qb,  M, N, K);
    gemm_bf16_kernel<unsigned short, false, false><<<ggrid, blk, 0, stream>>>(xb, wkb, nullptr, kb,  M, N, K);
    gemm_bf16_kernel<unsigned short, false, true ><<<ggrid, blk, 0, stream>>>(xb, wvb, nullptr, vtb, M, N, K);

    // Stage 2: attention
    attn_fwd_kernel<<<dim3(T / 128, 16, Bb), blk, 0, stream>>>(qb, kb, vtb, ab);

    // Stage 3: output projection + bias
    gemm_bf16_kernel<float, true, false><<<ggrid, blk, 0, stream>>>(ab, wob, bo, out, M, N, K);
}
